// QEffGrok1MoeBlock_52931176955914
// MI455X (gfx1250) — matrix operand order repaired and checked by reading, not compiled, not run on hardware
//
#include <hip/hip_runtime.h>
#include <hip/hip_bf16.h>
#include <math.h>

// ---------------- problem constants ----------------
#define Hdim 1024
#define Fdim 4096
#define Tdim 1024
#define Edim 8
#define NSLOT (2 * Tdim)   // total (token, expert) assignments = 2048

// LDS geometry (padded for bank-conflict-free b128 fragment loads)
#define A_STRIDE 40        // 32 K + pad, 80B rows (16B aligned)
#define B_STRIDE 24        // 16 elems + pad, 48B per-lane blocks (16B aligned)

typedef __attribute__((ext_vector_type(16))) __bf16 v16bf;
typedef __attribute__((ext_vector_type(8)))  __bf16 v8bf;
typedef __attribute__((ext_vector_type(4)))  __bf16 v4bf;
typedef __attribute__((ext_vector_type(2)))  __bf16 v2bf;
typedef __attribute__((ext_vector_type(8)))  float  v8f;

// ---------------- workspace layout ----------------
struct WSMeta {
    int   counts[Edim];
    int   cursors[Edim];
    int   expOff[Edim];
    int   eSel[NSLOT];
    int   slotSel[NSLOT];
    int   tokOf[NSLOT];
    float wSel[NSLOT];
};
#define GU_OFF   (65536)
#define GU_BYTES ((size_t)NSLOT * Fdim * 2)               // 16 MB
#define DOWN_OFF (GU_OFF + GU_BYTES)                      // f32, 8 MB

// ---------------- kernel 1: zero counters ----------------
__global__ void moe_init_kernel(WSMeta* meta) {
    int i = threadIdx.x;
    if (i < Edim) { meta->counts[i] = 0; meta->cursors[i] = 0; }
}

// ---------------- kernel 2: router ----------------
__global__ void moe_router_kernel(const float* __restrict__ x,
                                  const float* __restrict__ w_gate,
                                  float* __restrict__ logits_out,
                                  WSMeta* meta) {
    const int wave = threadIdx.x >> 5;
    const int lane = threadIdx.x & 31;
    const int t = blockIdx.x * 8 + wave;

    float acc[Edim];
#pragma unroll
    for (int e = 0; e < Edim; ++e) acc[e] = 0.0f;

    for (int h = lane; h < Hdim; h += 32) {
        float xv = x[(size_t)t * Hdim + h];
#pragma unroll
        for (int e = 0; e < Edim; ++e) acc[e] += xv * w_gate[h * Edim + e];
    }
#pragma unroll
    for (int e = 0; e < Edim; ++e) {
#pragma unroll
        for (int s = 16; s > 0; s >>= 1) acc[e] += __shfl_xor(acc[e], s, 32);
    }
    if (lane == 0) {
        float mx = acc[0];
#pragma unroll
        for (int e = 1; e < Edim; ++e) mx = fmaxf(mx, acc[e]);
        float p[Edim], sum = 0.0f;
#pragma unroll
        for (int e = 0; e < Edim; ++e) { p[e] = __expf(acc[e] - mx); sum += p[e]; }
        float inv = 1.0f / sum;
#pragma unroll
        for (int e = 0; e < Edim; ++e) p[e] *= inv;
        int e0 = 0;
#pragma unroll
        for (int e = 1; e < Edim; ++e) if (p[e] > p[e0]) e0 = e;
        int e1 = (e0 == 0) ? 1 : 0;
#pragma unroll
        for (int e = 0; e < Edim; ++e) if (e != e0 && p[e] > p[e1]) e1 = e;
#pragma unroll
        for (int e = 0; e < Edim; ++e) logits_out[t * Edim + e] = acc[e];
        meta->eSel[2 * t] = e0;  meta->eSel[2 * t + 1] = e1;
        meta->wSel[2 * t] = p[e0]; meta->wSel[2 * t + 1] = p[e1];
        atomicAdd(&meta->counts[e0], 1);
        atomicAdd(&meta->counts[e1], 1);
    }
}

// ---------------- kernel 3: exclusive scan ----------------
__global__ void moe_scan_kernel(WSMeta* meta) {
    if (threadIdx.x == 0) {
        int s = 0;
        for (int e = 0; e < Edim; ++e) { meta->expOff[e] = s; s += meta->counts[e]; }
    }
}

// ---------------- kernel 4: slot placement ----------------
__global__ void moe_place_kernel(WSMeta* meta) {
    int i = blockIdx.x * blockDim.x + threadIdx.x;
    int e = meta->eSel[i];
    int idx = atomicAdd(&meta->cursors[e], 1);
    int slot = meta->expOff[e] + idx;
    meta->slotSel[i] = slot;
    meta->tokOf[slot] = i >> 1;
}

// ---------------- WMMA fragment helpers ----------------
__device__ __forceinline__ v16bf frag_cat(v8bf lo, v8bf hi) {
    return __builtin_shufflevector(lo, hi, 0, 1, 2, 3, 4, 5, 6, 7,
                                           8, 9, 10, 11, 12, 13, 14, 15);
}
__device__ __forceinline__ v16bf load_fragA(const __bf16* lds, int mrow, int lane) {
    const __bf16* p = lds + mrow * A_STRIDE + ((lane >> 4) << 4);
    v8bf lo = *(const v8bf*)p;
    v8bf hi = *(const v8bf*)(p + 8);
    return frag_cat(lo, hi);
}
__device__ __forceinline__ v16bf load_fragB(const __bf16* lds, int tile, int lane) {
    const __bf16* p = lds + (tile * 32 + lane) * B_STRIDE;
    v8bf lo = *(const v8bf*)p;
    v8bf hi = *(const v8bf*)(p + 8);
    return frag_cat(lo, hi);
}
__device__ __forceinline__ v4bf cvt4(float4 f) {
    v4bf r; r[0] = (__bf16)f.x; r[1] = (__bf16)f.y; r[2] = (__bf16)f.z; r[3] = (__bf16)f.w;
    return r;
}

// B-writer geometry: thread owns rows (kkB, kkB+1) x 4 cols ff0..ff0+3.
// Element (kk, ff) lives at [tile=ff>>4][lane=(ff&15)+16*(kk>>4)][i=kk&15];
// the (kk,kk+1) pair is contiguous (i even) -> ds_store_b32 per column.
__device__ __forceinline__ void b_pair_idx(int kkB, int ff0, int* bIdx) {
#pragma unroll
    for (int j = 0; j < 4; ++j) {
        int ff = ff0 + j;
        int lf = (ff & 15) + ((kkB >> 4) << 4);
        bIdx[j] = (((ff >> 4) * 32) + lf) * B_STRIDE + (kkB & 15);
    }
}

// ---------------- kernel 5: grouped gate/up GEMM + exact GELU ----------------
// block = 256 threads = 8 waves (2 M x 4 N); tile = 32 tokens x 64 F; double-buffered
__global__ void moe_gateup_kernel(const float* __restrict__ x,
                                  const float* __restrict__ w_g,
                                  const float* __restrict__ w_v,
                                  WSMeta* meta, __bf16* __restrict__ gu) {
    const int e = blockIdx.z;
    const int cnt = meta->counts[e];
    const int mTile = blockIdx.y;
    if (mTile * 32 >= cnt) return;
    const int fBase = blockIdx.x * 64;
    const int off = meta->expOff[e];

    __shared__ __bf16 ldsA[2][32 * A_STRIDE];
    __shared__ __bf16 ldsG[2][4 * 32 * B_STRIDE];
    __shared__ __bf16 ldsV[2][4 * 32 * B_STRIDE];
    __shared__ int toks[32];

    const int tid = threadIdx.x;
    const int lane = tid & 31, wave = tid >> 5;
    const int wM = wave >> 2, wN = wave & 3;

    if (tid < 32) {
        int idx = mTile * 32 + tid;
        toks[tid] = meta->tokOf[off + ((idx < cnt) ? idx : 0)];
    }
    __syncthreads();

    // staging geometry (all loop-invariant)
    const int mmA = tid >> 3, kk0A = (tid & 7) * 4;       // A: 1 float4 / thread
    const int myTok = toks[mmA];
    const int kkB = (tid >> 4) * 2;                       // B: row-pair
    const int ff0 = (tid & 15) * 4;                       // B: 4 columns
    int bIdx[4];
    b_pair_idx(kkB, ff0, bIdx);

    const float* xA  = x + (size_t)myTok * Hdim + kk0A;
    const float* wgp = w_g + (size_t)e * Hdim * Fdim + (size_t)kkB * Fdim + fBase + ff0;
    const float* wvp = w_v + (size_t)e * Hdim * Fdim + (size_t)kkB * Fdim + fBase + ff0;

    // prologue: stage k0 = 0 into registers
    float4 aR  = *(const float4*)(xA);
    float4 g0R = *(const float4*)(wgp);
    float4 g1R = *(const float4*)(wgp + Fdim);
    float4 v0R = *(const float4*)(wvp);
    float4 v1R = *(const float4*)(wvp + Fdim);

    v8f accG = {}; v8f accV = {};
    int cur = 0;
    for (int k0 = 0; k0 < Hdim; k0 += 32) {
        // registers -> LDS[cur]
        *(v4bf*)(&ldsA[cur][mmA * A_STRIDE + kk0A]) = cvt4(aR);
#pragma unroll
        for (int j = 0; j < 4; ++j) {
            v2bf pg; pg[0] = (__bf16)((&g0R.x)[j]); pg[1] = (__bf16)((&g1R.x)[j]);
            v2bf pv; pv[0] = (__bf16)((&v0R.x)[j]); pv[1] = (__bf16)((&v1R.x)[j]);
            *(v2bf*)(&ldsG[cur][bIdx[j]]) = pg;
            *(v2bf*)(&ldsV[cur][bIdx[j]]) = pv;
        }
        __syncthreads();
        // issue next tile's global loads before consuming LDS (overlap with WMMA)
        if (k0 + 32 < Hdim) {
            const size_t ko = (size_t)(k0 + 32);
            aR  = *(const float4*)(xA + ko);
            g0R = *(const float4*)(wgp + ko * Fdim);
            g1R = *(const float4*)(wgp + (ko + 1) * Fdim);
            v0R = *(const float4*)(wvp + ko * Fdim);
            v1R = *(const float4*)(wvp + (ko + 1) * Fdim);
        }
        v16bf a  = load_fragA(ldsA[cur], (lane & 15) + wM * 16, lane);
        v16bf bg = load_fragB(ldsG[cur], wN, lane);
        v16bf bv = load_fragB(ldsV[cur], wN, lane);
        accG = __builtin_amdgcn_wmma_f32_16x16x32_bf16(false, a, false, bg,
                                                       (short)0, accG, false, false);
        accV = __builtin_amdgcn_wmma_f32_16x16x32_bf16(false, a, false, bv,
                                                       (short)0, accV, false, false);
        cur ^= 1;
    }

    const int rbase = (lane >> 4) * 8;
    const int col = lane & 15;
#pragma unroll
    for (int r = 0; r < 8; ++r) {
        int row = wM * 16 + r + rbase;
        int idx = mTile * 32 + row;
        if (idx < cnt) {
            float g = accG[r], u = accV[r];
            float act = 0.5f * g * (1.0f + erff(g * 0.70710678118654752f)) * u;
            gu[(size_t)(off + idx) * Fdim + fBase + wN * 16 + col] = (__bf16)act;
        }
    }
}

// ---------------- kernel 6: grouped down GEMM ----------------
__global__ void moe_down_kernel(const __bf16* __restrict__ gu,
                                const float* __restrict__ w_1,
                                WSMeta* meta, float* __restrict__ down) {
    const int e = blockIdx.z;
    const int cnt = meta->counts[e];
    const int mTile = blockIdx.y;
    if (mTile * 32 >= cnt) return;
    const int hBase = blockIdx.x * 64;
    const int off = meta->expOff[e];

    __shared__ __bf16 ldsA[2][32 * A_STRIDE];
    __shared__ __bf16 ldsB[2][4 * 32 * B_STRIDE];

    const int tid = threadIdx.x;
    const int lane = tid & 31, wave = tid >> 5;
    const int wM = wave >> 2, wN = wave & 3;

    const int mmA = tid >> 3, kk0A = (tid & 7) * 4;       // A: 4 bf16 (8B) / thread
    const int idxA = mTile * 32 + mmA;
    const __bf16* gA = gu + (size_t)(off + ((idxA < cnt) ? idxA : 0)) * Fdim + kk0A;

    const int kkB = (tid >> 4) * 2;
    const int ff0 = (tid & 15) * 4;
    int bIdx[4];
    b_pair_idx(kkB, ff0, bIdx);
    const float* w1p = w_1 + (size_t)e * Fdim * Hdim + (size_t)kkB * Hdim + hBase + ff0;

    uint2  aR = *(const uint2*)(gA);
    float4 b0R = *(const float4*)(w1p);
    float4 b1R = *(const float4*)(w1p + Hdim);

    v8f acc = {};
    int cur = 0;
    for (int k0 = 0; k0 < Fdim; k0 += 32) {
        *(uint2*)(&ldsA[cur][mmA * A_STRIDE + kk0A]) = aR;
#pragma unroll
        for (int j = 0; j < 4; ++j) {
            v2bf pb; pb[0] = (__bf16)((&b0R.x)[j]); pb[1] = (__bf16)((&b1R.x)[j]);
            *(v2bf*)(&ldsB[cur][bIdx[j]]) = pb;
        }
        __syncthreads();
        if (k0 + 32 < Fdim) {
            const size_t ko = (size_t)(k0 + 32);
            aR  = *(const uint2*)(gA + ko);
            b0R = *(const float4*)(w1p + ko * Hdim);
            b1R = *(const float4*)(w1p + (ko + 1) * Hdim);
        }
        v16bf a = load_fragA(ldsA[cur], (lane & 15) + wM * 16, lane);
        v16bf b = load_fragB(ldsB[cur], wN, lane);
        acc = __builtin_amdgcn_wmma_f32_16x16x32_bf16(false, a, false, b,
                                                      (short)0, acc, false, false);
        cur ^= 1;
    }

    const int rbase = (lane >> 4) * 8;
    const int col = lane & 15;
#pragma unroll
    for (int r = 0; r < 8; ++r) {
        int row = wM * 16 + r + rbase;
        int idx = mTile * 32 + row;
        if (idx < cnt)
            down[(size_t)(off + idx) * Hdim + hBase + wN * 16 + col] = acc[r];
    }
}

// ---------------- kernel 7: weighted combine ----------------
__global__ void moe_combine_kernel(const float* __restrict__ down,
                                   const WSMeta* __restrict__ meta,
                                   float* __restrict__ out) {
    int gid = blockIdx.x * blockDim.x + threadIdx.x;
    int t = gid >> 10;
    int h = gid & (Hdim - 1);
    int s0 = meta->slotSel[2 * t], s1 = meta->slotSel[2 * t + 1];
    float w0 = meta->wSel[2 * t], w1 = meta->wSel[2 * t + 1];
    out[gid] = w0 * down[(size_t)s0 * Hdim + h] + w1 * down[(size_t)s1 * Hdim + h];
}

// ---------------- launcher ----------------
extern "C" void kernel_launch(void* const* d_in, const int* in_sizes, int n_in,
                              void* d_out, int out_size, void* d_ws, size_t ws_size,
                              hipStream_t stream) {
    const float* x      = (const float*)d_in[0];
    const float* w_gate = (const float*)d_in[1];
    const float* w_g    = (const float*)d_in[2];
    const float* w_v    = (const float*)d_in[3];
    const float* w_1    = (const float*)d_in[4];

    float* out    = (float*)d_out;
    float* logits = out + (size_t)Tdim * Hdim;

    WSMeta* meta = (WSMeta*)d_ws;
    __bf16* gu   = (__bf16*)((char*)d_ws + GU_OFF);
    float*  down = (float*)((char*)d_ws + DOWN_OFF);

    moe_init_kernel<<<1, 32, 0, stream>>>(meta);
    moe_router_kernel<<<Tdim / 8, 256, 0, stream>>>(x, w_gate, logits, meta);
    moe_scan_kernel<<<1, 32, 0, stream>>>(meta);
    moe_place_kernel<<<NSLOT / 256, 256, 0, stream>>>(meta);
    moe_gateup_kernel<<<dim3(Fdim / 64, Tdim / 32, Edim), 256, 0, stream>>>(
        x, w_g, w_v, meta, gu);
    moe_down_kernel<<<dim3(Hdim / 64, Tdim / 32, Edim), 256, 0, stream>>>(
        gu, w_1, meta, down);
    moe_combine_kernel<<<(Tdim * Hdim) / 256, 256, 0, stream>>>(down, meta, out);
}
